// TriangleAttention_63763084476751
// MI455X (gfx1250) — matrix operand order repaired and checked by reading, not compiled, not run on hardware
//
#include <hip/hip_runtime.h>
#include <hip/hip_bf16.h>

#define NRES 256
#define CZ 128
#define NHEAD 4
#define CH 32
#define NPAIR (NRES * NRES)

typedef __attribute__((ext_vector_type(16))) _Float16 v16h;
typedef __attribute__((ext_vector_type(8)))  _Float16 v8h;
typedef __attribute__((ext_vector_type(8)))  float    v8f;

// ---- WMMA tile loaders (wave32, layouts per CDNA5 ISA 7.12.2) ----

// A-matrix 16x32 f16: lanes 0-15 hold M=lane, K={0..7,16..23}; lanes 16-31 hold
// M=lane-16, K={8..15,24..31}.  Two contiguous 16B loads per lane.
__device__ __forceinline__ v16h wmma_load_A(const _Float16* base, int ld, int lane) {
  const int half = (lane >> 4) & 1;
  const int m = lane & 15;
  const _Float16* p = base + m * ld + half * 8;
  v8h a0 = *(const v8h*)(p);
  v8h a1 = *(const v8h*)(p + 16);
  v16h a;
#pragma unroll
  for (int t = 0; t < 8; ++t) { a[t] = a0[t]; a[8 + t] = a1[t]; }
  return a;
}

// B-matrix 32x16 f16 where B[k][n] = src[n*ld + k] (column n of B = contiguous
// row n of src). lanes 0-15: N=lane, K={0..7,16..23}; lanes 16-31: N=lane-16, K={8..15,24..31}.
__device__ __forceinline__ v16h wmma_load_B_T(const _Float16* base, int ld, int lane) {
  const int hk = (lane >> 4) & 1;
  const int n = lane & 15;
  const _Float16* p = base + n * ld + hk * 8;
  v8h b0 = *(const v8h*)(p);
  v8h b1 = *(const v8h*)(p + 16);
  v16h b;
#pragma unroll
  for (int t = 0; t < 8; ++t) { b[t] = b0[t]; b[8 + t] = b1[t]; }
  return b;
}

__device__ __forceinline__ v8f wmma_f16(v16h a, v16h b, v8f c) {
  return __builtin_amdgcn_wmma_f32_16x16x32_f16(false, a, false, b, (short)0, c,
                                                false, false);
}

__device__ __forceinline__ float fast_rcp(float x) {
  return __builtin_amdgcn_rcpf(x);   // v_rcp_f32, no div fixup chain
}

// ---- Kernel 0: convert + transpose weights to f16 ----
// dst layout: [5][out=128][in=128], slots: q,k,v,g,o
__global__ void convert_weights_kernel(const float* __restrict__ wq, const float* __restrict__ wk,
                                       const float* __restrict__ wv, const float* __restrict__ wg,
                                       const float* __restrict__ wo, _Float16* __restrict__ dst) {
  int idx = blockIdx.x * blockDim.x + threadIdx.x;   // 0 .. 5*16384-1
  int w  = idx >> 14;
  int o  = (idx >> 7) & 127;
  int in = idx & 127;
  const float* src = (w == 0) ? wq : (w == 1) ? wk : (w == 2) ? wv : (w == 3) ? wg : wo;
  dst[idx] = (_Float16)src[in * 128 + o];
}

// ---- Kernel 1: LayerNorm + q/k/v/g projections (WMMA) + pair-bias projection ----
// V is stored transposed per (i,h) slice: vT[((i*NHEAD+h)*CH + c)*NRES + j]
__global__ __launch_bounds__(128)
void prep_kernel(const float* __restrict__ z, const float* __restrict__ gamma,
                 const float* __restrict__ beta, const float* __restrict__ wbias,
                 const float* __restrict__ bg, const _Float16* __restrict__ w16,
                 _Float16* __restrict__ qh, _Float16* __restrict__ kh,
                 _Float16* __restrict__ vT, _Float16* __restrict__ gh,
                 float* __restrict__ biasb) {
  __shared__ _Float16 zn[16][CZ];
  __shared__ float red[16][8][2];
  const int t = threadIdx.x;
  const int rowbase = blockIdx.x * 16;
  const int row = t >> 3, sub = t & 7, c0 = sub * 16;

  const int wid = t >> 5, lane = t & 31;
  const _Float16* wT = w16 + (size_t)wid * CZ * CZ;   // [out][in]
  __builtin_prefetch(wT + (size_t)lane * 512, 0, 1);  // pull weight panel toward WGP

  // ---- LayerNorm over the channel dim (biased variance, eps=1e-5) ----
  const float* zr = z + (size_t)(rowbase + row) * CZ;
  float zv[16];
  float s = 0.f, s2 = 0.f;
#pragma unroll
  for (int c = 0; c < 16; ++c) { float v = zr[c0 + c]; zv[c] = v; s += v; s2 += v * v; }
  red[row][sub][0] = s; red[row][sub][1] = s2;
  __syncthreads();
  float ts = 0.f, ts2 = 0.f;
#pragma unroll
  for (int p = 0; p < 8; ++p) { ts += red[row][p][0]; ts2 += red[row][p][1]; }
  const float mu   = ts * (1.f / CZ);
  const float var  = ts2 * (1.f / CZ) - mu * mu;
  const float rstd = rsqrtf(var + 1e-5f);
#pragma unroll
  for (int c = 0; c < 16; ++c)
    zn[row][c0 + c] = (_Float16)((zv[c] - mu) * rstd * gamma[c0 + c] + beta[c0 + c]);
  __syncthreads();

  // ---- pair bias: bias[h][p] = sum_c zn[p][c] * wbias[c][h] ----
  if (t < 64) {
    const int br = t >> 2, hh = t & 3;
    float acc = 0.f;
#pragma unroll 4
    for (int c = 0; c < CZ; ++c) acc += (float)zn[br][c] * wbias[c * NHEAD + hh];
    biasb[(size_t)hh * NPAIR + rowbase + br] = acc;
  }

  // ---- projections: wave w computes zn @ W_w  (w: 0=q,1=k,2=v,3=g) ----
  const int half = (lane >> 4) & 1, n = lane & 15;

  // hoist the 4 A-tiles (16x32 each, spanning the 128 input channels)
  v16h Az[4];
#pragma unroll
  for (int kc = 0; kc < 4; ++kc) Az[kc] = wmma_load_A(&zn[0][kc * 32], CZ, lane);

#pragma unroll
  for (int nt = 0; nt < 8; ++nt) {
    v8f acc = {};
#pragma unroll
    for (int kc = 0; kc < 4; ++kc) {
      v16h B = wmma_load_B_T(wT + (size_t)(nt * 16) * CZ + kc * 32, CZ, lane);
      acc = wmma_f16(Az[kc], B, acc);
    }
    const int col = nt * 16 + n;
    if (wid == 3) {                       // gate: sigmoid(zn@wg + bg)
      const float bgv = bg[col];
#pragma unroll
      for (int r = 0; r < 8; ++r) {
        const int rw = rowbase + r + half * 8;
        float x = acc[r] + bgv;
        gh[(size_t)rw * CZ + col] = (_Float16)fast_rcp(1.f + __expf(-x));
      }
    } else if (wid == 2) {                // V: store transposed per (i,h) slice
#pragma unroll
      for (int r = 0; r < 8; ++r) {
        const int p = rowbase + r + half * 8;
        const int i = p >> 8, j = p & 255;
        // (h*CH + c) == col, NHEAD*CH == 128
        vT[((size_t)i * 128 + col) * NRES + j] = (_Float16)acc[r];
      }
    } else {
      _Float16* dst = (wid == 0) ? qh : kh;
#pragma unroll
      for (int r = 0; r < 8; ++r) {
        const int rw = rowbase + r + half * 8;
        dst[(size_t)rw * CZ + col] = (_Float16)acc[r];
      }
    }
  }
}

// ---- Kernel 2: per-(i,h) attention. One wave owns a 16-row tile of S. ----
__global__ __launch_bounds__(128)
void attn_kernel(const _Float16* __restrict__ qh, const _Float16* __restrict__ kh,
                 const _Float16* __restrict__ vT, const _Float16* __restrict__ gh,
                 const float* __restrict__ biasb, _Float16* __restrict__ goh) {
  __shared__ _Float16 sS[4][16][NRES];   // 32 KB: per-wave softmax tiles
  const int t = threadIdx.x, w = t >> 5, lane = t & 31;
  const int i = blockIdx.z, h = blockIdx.y;
  const int j0 = (blockIdx.x * 4 + w) * 16;
  const int half = (lane >> 4) & 1, n = lane & 15;
  const float scale = 0.1767766952966369f;   // 1/sqrt(32)

  // Q tile (16 rows x 32 ch), single A operand (K==CH==32)
  const _Float16* Qb = qh + (size_t)(i * NRES + j0) * CZ + h * CH;
  v16h A = wmma_load_A(Qb, CZ, lane);

  // S = Q K^T : 16 tiles of 16x16, fp32 accumulators stay in VGPRs (C layout)
  v8f acc[16];
#pragma unroll
  for (int jt = 0; jt < 16; ++jt) {
    const _Float16* Kb = kh + (size_t)(i * NRES + jt * 16) * CZ + h * CH;
    v16h B = wmma_load_B_T(Kb, CZ, lane);   // B[k][n] = K[row n][ch k]
    v8f c = {};
    acc[jt] = wmma_f16(A, B, c);
  }

  // scale + pair bias (bias indexed by (j,k), independent of i)
  const float* bb = biasb + (size_t)h * NPAIR;
#pragma unroll
  for (int jt = 0; jt < 16; ++jt) {
    const int k = jt * 16 + n;
#pragma unroll
    for (int r = 0; r < 8; ++r) {
      const int j = j0 + r + half * 8;
      acc[jt][r] = acc[jt][r] * scale + bb[(size_t)j * NRES + k];
    }
  }

  // row softmax: C layout => row (r + half*8) lives in VGPR slot r across a
  // 16-lane half; reduce over the 16 tiles per lane, then shfl_xor across 16.
  float rinv[8];
#pragma unroll
  for (int r = 0; r < 8; ++r) {
    float m = -3.0e38f;
#pragma unroll
    for (int jt = 0; jt < 16; ++jt) m = fmaxf(m, acc[jt][r]);
#pragma unroll
    for (int off = 1; off < 16; off <<= 1) m = fmaxf(m, __shfl_xor(m, off, 32));
    float su = 0.f;
#pragma unroll
    for (int jt = 0; jt < 16; ++jt) {
      float e = __expf(acc[jt][r] - m);
      acc[jt][r] = e;
      su += e;
    }
#pragma unroll
    for (int off = 1; off < 16; off <<= 1) su += __shfl_xor(su, off, 32);
    rinv[r] = fast_rcp(su);
  }

  // spill normalized probs to LDS as f16 so they can be re-read in A layout
#pragma unroll
  for (int jt = 0; jt < 16; ++jt) {
    const int col = jt * 16 + n;
#pragma unroll
    for (int r = 0; r < 8; ++r)
      sS[w][r + half * 8][col] = (_Float16)(acc[jt][r] * rinv[r]);
  }
  __syncthreads();

  // O = P @ V (K=256 in 8 chunks of 32); probs re-read once into registers
  v16h As[8];
#pragma unroll
  for (int kt = 0; kt < 8; ++kt) As[kt] = wmma_load_A(&sS[w][0][kt * 32], NRES, lane);

  const _Float16* vTb = vT + ((size_t)i * 128 + (size_t)h * CH) * NRES;
#pragma unroll
  for (int ct = 0; ct < 2; ++ct) {
    v8f o = {};
#pragma unroll
    for (int kt = 0; kt < 8; ++kt) {
      // B[k][n] = V[j=kt*32+k][c=ct*16+n] = vT[(ct*16+n)*NRES + kt*32 + k]
      v16h Bv = wmma_load_B_T(vTb + (size_t)(ct * 16) * NRES + kt * 32, NRES, lane);
      o = wmma_f16(As[kt], Bv, o);
    }
#pragma unroll
    for (int r = 0; r < 8; ++r) {
      const int j = j0 + r + half * 8;
      const size_t idx = (size_t)(i * NRES + j) * CZ + h * CH + ct * 16 + n;
      float gv = (float)gh[idx];
      goh[idx] = (_Float16)(o[r] * gv);
    }
  }
}

// ---- Kernel 3: out = (g*o) @ wo + bo, fp32 output ----
__global__ __launch_bounds__(128)
void outproj_kernel(const _Float16* __restrict__ goh, const _Float16* __restrict__ w16,
                    const float* __restrict__ bo, float* __restrict__ out) {
  const int t = threadIdx.x, w = t >> 5, lane = t & 31;
  const int rowbase = blockIdx.x * 16;
  const int half = (lane >> 4) & 1, n = lane & 15;
  const _Float16* woT = w16 + (size_t)4 * CZ * CZ;   // slot 4 = wo transposed
  __builtin_prefetch(woT + (size_t)lane * 512, 0, 1);

  // hoist A tiles (shared across both N-tiles of this wave)
  v16h Ar[4];
#pragma unroll
  for (int kc = 0; kc < 4; ++kc)
    Ar[kc] = wmma_load_A(goh + (size_t)rowbase * CZ + kc * 32, CZ, lane);

#pragma unroll
  for (int nn = 0; nn < 2; ++nn) {
    const int nt = w * 2 + nn;
    v8f acc = {};
#pragma unroll
    for (int kc = 0; kc < 4; ++kc) {
      v16h B = wmma_load_B_T(woT + (size_t)(nt * 16) * CZ + kc * 32, CZ, lane);
      acc = wmma_f16(Ar[kc], B, acc);
    }
    const int col = nt * 16 + n;
    const float bov = bo[col];
#pragma unroll
    for (int r = 0; r < 8; ++r)
      out[(size_t)(rowbase + r + half * 8) * CZ + col] = acc[r] + bov;
  }
}

extern "C" void kernel_launch(void* const* d_in, const int* in_sizes, int n_in,
                              void* d_out, int out_size, void* d_ws, size_t ws_size,
                              hipStream_t stream) {
  const float* z     = (const float*)d_in[0];
  const float* gamma = (const float*)d_in[1];
  const float* beta  = (const float*)d_in[2];
  const float* wbias = (const float*)d_in[3];
  const float* wq    = (const float*)d_in[4];
  const float* wk    = (const float*)d_in[5];
  const float* wv    = (const float*)d_in[6];
  const float* wg    = (const float*)d_in[7];
  const float* bg    = (const float*)d_in[8];
  const float* wo    = (const float*)d_in[9];
  const float* bo    = (const float*)d_in[10];
  float* out = (float*)d_out;

  // workspace layout
  char* ws = (char*)d_ws;
  const size_t HSZ = (size_t)NPAIR * CZ * sizeof(_Float16);   // 16 MB per f16 tensor
  _Float16* w16  = (_Float16*)ws;                  // 5*128*128 f16 = 160 KB
  _Float16* qh   = (_Float16*)(ws + (1u << 18));   // +256 KB
  _Float16* kh   = (_Float16*)((char*)qh + HSZ);
  _Float16* vT   = (_Float16*)((char*)kh + HSZ);   // transposed V
  _Float16* gh   = (_Float16*)((char*)vT + HSZ);
  _Float16* goh  = (_Float16*)((char*)gh + HSZ);
  float*    biasb = (float*)((char*)goh + HSZ);    // 4*65536 f32 = 1 MB

  convert_weights_kernel<<<(5 * 16384) / 256, 256, 0, stream>>>(wq, wk, wv, wg, wo, w16);
  prep_kernel<<<NPAIR / 16, 128, 0, stream>>>(z, gamma, beta, wbias, bg, w16,
                                              qh, kh, vT, gh, biasb);
  attn_kernel<<<dim3(NRES / 64, NHEAD, NRES), 128, 0, stream>>>(qh, kh, vT, gh,
                                                                biasb, goh);
  outproj_kernel<<<NPAIR / 16, 128, 0, stream>>>(goh, w16, bo, out);
}
